// BiMPMMatching_83537113907966
// MI455X (gfx1250) — matrix operand order repaired and checked by reading, not compiled, not run on hardware
//
#include <hip/hip_runtime.h>

// ---------------------------------------------------------------- constants
constexpr int   S    = 256;   // S1 == S2
constexpr int   H    = 100;
constexpr int   BB   = 8;     // batch
constexpr int   P    = 20;    // perspectives
constexpr int   HPAD = 128;   // H padded to WMMA K multiple (32)
constexpr int   OD   = 105;   // output feature dim
constexpr float EPSV = 1e-8f;
constexpr float MINV = -1e7f;

typedef __bf16 bf16_t;
typedef __bf16 v16bf __attribute__((ext_vector_type(16)));
typedef __bf16 v8bf  __attribute__((ext_vector_type(8)));
typedef float  v8f   __attribute__((ext_vector_type(8)));

union V16U { v16bf v; v8bf h[2]; };

static __device__ __forceinline__ v8f wmma_bf16(v16bf a, v16bf b, v8f c) {
  // D(16x16,f32) = A(16x32,bf16) x B(32x16,bf16) + C
  return __builtin_amdgcn_wmma_f32_16x16x32_bf16(
      /*neg_a=*/false, a, /*neg_b=*/false, b,
      /*c_mod=*/(short)0, c, /*reuse_a=*/false, /*reuse_b=*/false);
}

// Fast reciprocal (v_rcp_f32), used instead of IEEE divides in the k_pw and
// k_mpm_final epilogues.
static __device__ __forceinline__ float frcp(float x) {
  return __builtin_amdgcn_rcpf(x);
}

// A operand: row-major [row][k]; lane m=L&15 -> row, hl=L>>4.
// elems 0..7 = K[k0+8*hl .. +7], elems 8..15 = K[k0+16+8*hl .. +7]
static __device__ __forceinline__ v16bf load_a(const bf16_t* base, int ld,
                                               int row, int k0, int hl) {
  const bf16_t* p = base + row * ld + k0 + 8 * hl;
  V16U u;
  u.h[0] = *(const v8bf*)p;
  u.h[1] = *(const v8bf*)(p + 16);
  return u.v;
}
// B operand: B[k][n] = rowmajor[n][k]; lane n=L&15 -> row,
// elems 0..7 = K[k0+8*hl .. +7], elems 8..15 = K[k0+16+8*hl .. +7]
static __device__ __forceinline__ v16bf load_b(const bf16_t* base, int ld,
                                               int row, int k0, int hl) {
  const bf16_t* p = base + row * ld + k0 + 8 * hl;
  V16U u;
  u.h[0] = *(const v8bf*)p;
  u.h[1] = *(const v8bf*)(p + 16);
  return u.v;
}

// ---------------------------------------------------------------- prep
// grid(S, B, 2), block 128. Mask contexts, f32 + bf16 (padded) + bf16-transposed
// copies, per-row reciprocal L2 norms (1 exact divide per row, hoisted out of
// the O(S^2) cos epilogue).
__global__ void k_prep(const float* ctx_p, const int* mask_p,
                       const float* ctx_h, const int* mask_h,
                       float* cp, float* ch, bf16_t* cp16, bf16_t* ch16,
                       bf16_t* cpT16, bf16_t* chT16,
                       float* rnorm_p, float* rnorm_h) {
  int s = blockIdx.x, b = blockIdx.y, side = blockIdx.z, t = threadIdx.x;
  const float* src = (side ? ctx_h : ctx_p) + (b * S + s) * H;
  const int*   mk  = side ? mask_h : mask_p;
  float m = (float)mk[b * S + s];
  float v = (t < H) ? src[t] * m : 0.f;
  float*  dst = side ? ch : cp;
  bf16_t* d16 = side ? ch16 : cp16;
  bf16_t* dT  = side ? chT16 : cpT16;
  if (t < H) dst[(b * S + s) * H + t] = v;
  d16[(b * S + s) * HPAD + t] = (bf16_t)v;
  dT[(b * HPAD + t) * S + s]  = (bf16_t)v;
  __shared__ float red[128];
  red[t] = v * v;
  __syncthreads();
  for (int w = 64; w > 0; w >>= 1) {
    if (t < w) red[t] += red[t + w];
    __syncthreads();
  }
  if (t == 0)
    (side ? rnorm_h : rnorm_p)[b * S + s] = 1.f / fmaxf(sqrtf(red[0]), EPSV);
}

// grid(B,2), block 256. Sequence lengths + last valid (masked) token row.
__global__ void k_lens_last(const int* mask_p, const int* mask_h,
                            const float* cp, const float* ch,
                            float* len_p, float* len_h,
                            float* last_p, float* last_h) {
  int b = blockIdx.x, side = blockIdx.y, t = threadIdx.x;
  const int* mk = (side ? mask_h : mask_p) + b * S;
  __shared__ int red[256];
  __shared__ int slen;
  red[t] = mk[t];
  __syncthreads();
  for (int w = 128; w > 0; w >>= 1) {
    if (t < w) red[t] += red[t + w];
    __syncthreads();
  }
  if (t == 0) { slen = red[0]; (side ? len_h : len_p)[b] = (float)red[0]; }
  __syncthreads();
  int idx = slen > 0 ? slen - 1 : 0;
  const float* src = (side ? ch : cp) + (b * S + idx) * H;
  if (t < H) (side ? last_h : last_p)[b * H + t] = src[t];
}

// grid(P), block 128. Squared maxpool weights, zero-padded to HPAD.
__global__ void k_w2(const float* w, float* w2) {
  int p = blockIdx.x, t = threadIdx.x;
  float x = (t < H) ? w[p * H + t] : 0.f;
  w2[p * HPAD + t] = x * x;
}

// grid(S, B, 2), block 32. nw[b][p][s] = sqrt(sum_h w2[p][h] * v[s][h]^2)
__global__ void k_wnorm(const float* cp, const float* ch, const float* w2,
                        float* nwp, float* nwh) {
  int s = blockIdx.x, b = blockIdx.y, side = blockIdx.z, t = threadIdx.x;
  const float* v = (side ? ch : cp) + (b * S + s) * H;
  __shared__ float sv[H];
  for (int u = t; u < H; u += 32) sv[u] = v[u] * v[u];
  __syncthreads();
  if (t < P) {
    float acc = 0.f;
    for (int h = 0; h < H; ++h) acc += w2[t * HPAD + h] * sv[h];
    (side ? nwh : nwp)[(b * P + t) * S + s] = sqrtf(acc);
  }
}

// ---------------------------------------------------------------- cos GEMM
// grid(S/16, S/16, B), block 32 (one wave -> one 16x16 tile).
// Epilogue: two multiplies per element (reciprocal norms precomputed).
__global__ void k_cos_gemm(const bf16_t* cp16, const bf16_t* ch16,
                           const float* rnorm_p, const float* rnorm_h,
                           float* cosm) {
  int jt = blockIdx.x, it = blockIdx.y, b = blockIdx.z;
  int lane = threadIdx.x, m = lane & 15, hl = lane >> 4;
  int i0 = it * 16, j0 = jt * 16;
  const bf16_t* A  = cp16 + (size_t)b * S * HPAD;
  const bf16_t* Bm = ch16 + (size_t)b * S * HPAD;
  v8f c = {0.f, 0.f, 0.f, 0.f, 0.f, 0.f, 0.f, 0.f};
  for (int k0 = 0; k0 < HPAD; k0 += 32) {
    v16bf a  = load_a(A, HPAD, i0 + m, k0, hl);
    v16bf bb = load_b(Bm, HPAD, j0 + m, k0, hl);
    c = wmma_bf16(a, bb, c);
  }
  int j = j0 + m;
  float rh = rnorm_h[b * S + j];
#pragma unroll
  for (int r = 0; r < 8; ++r) {
    int i = i0 + r + 8 * hl;
    cosm[((size_t)b * S + i) * S + j] = c[r] * rnorm_p[b * S + i] * rh;
  }
}

// grid(S, B), block 256. Row masked-max/mean over j; emit bf16 att + attT;
// reciprocal rowsum (attention normalizer, hoisted out of attmean epilogue).
__global__ void k_reduce_rows(const float* cosm, const int* mask_h,
                              const float* len_h, bf16_t* att16, bf16_t* attT16,
                              float* rrowsum, float* out) {
  int i = blockIdx.x, b = blockIdx.y, j = threadIdx.x;
  float v = cosm[((size_t)b * S + i) * S + j];
  att16[((size_t)b * S + i) * S + j]  = (bf16_t)v;
  attT16[((size_t)b * S + j) * S + i] = (bf16_t)v;
  float msk = (float)mask_h[b * S + j];
  __shared__ float smax[256], ssum[256];
  smax[j] = v + (1.f - msk) * MINV;
  ssum[j] = v;  // masked cols are exactly 0 -> unmasked sum == masked sum
  __syncthreads();
  for (int w = 128; w > 0; w >>= 1) {
    if (j < w) { smax[j] = fmaxf(smax[j], smax[j + w]); ssum[j] += ssum[j + w]; }
    __syncthreads();
  }
  if (j == 0) {
    rrowsum[b * S + i] = 1.f / fmaxf(ssum[0], EPSV);
    float* o = out + ((size_t)(0 * BB + b) * S + i) * OD;
    o[0] = smax[0];
    o[1] = ssum[0] / fmaxf(len_h[b], EPSV);
  }
}

// grid(S, B), block 256. Column masked-max/mean over i; reciprocal colsum.
__global__ void k_reduce_cols(const float* cosm, const int* mask_p,
                              const float* len_p, float* rcolsum, float* out) {
  int j = blockIdx.x, b = blockIdx.y, i = threadIdx.x;
  float v = cosm[((size_t)b * S + i) * S + j];
  float msk = (float)mask_p[b * S + i];
  __shared__ float smax[256], ssum[256];
  smax[i] = v + (1.f - msk) * MINV;
  ssum[i] = v;
  __syncthreads();
  for (int w = 128; w > 0; w >>= 1) {
    if (i < w) { smax[i] = fmaxf(smax[i], smax[i + w]); ssum[i] += ssum[i + w]; }
    __syncthreads();
  }
  if (i == 0) {
    rcolsum[b * S + j] = 1.f / fmaxf(ssum[0], EPSV);
    float* o = out + ((size_t)(1 * BB + b) * S + j) * OD;
    o[0] = smax[0];
    o[1] = ssum[0] / fmaxf(len_p[b], EPSV);
  }
}

// ---------------------------------------------------------------- maxpool pw
// Fused per-perspective pairwise-cosine GEMM + masked max/mean reduction.
// grid(S/16, P, B), block 32. A side scaled by w^2 on the fly (f32 source),
// B side is the shared bf16 context. The eps applies to the norm *product*
// (per the reference), so it cannot be hoisted -> single v_rcp_f32 instead of
// an IEEE divide chain in the per-element epilogue.
__global__ void k_pw(const float* v1f, const bf16_t* v2_16, const float* w2,
                     const float* n1, const float* n2, const int* mask2,
                     const float* len2, float* out, int outSide) {
  int it = blockIdx.x, p = blockIdx.y, b = blockIdx.z;
  int lane = threadIdx.x, m = lane & 15, hl = lane >> 4;
  int i0 = it * 16;
  const float* row = v1f + ((size_t)b * S + i0 + m) * H;
  const float* wp  = w2 + p * HPAD;
  v16bf a4[4];
#pragma unroll
  for (int kk = 0; kk < 4; ++kk) {
    int k0 = kk * 32;
    V16U u;
#pragma unroll
    for (int e = 0; e < 8; ++e) {
      int ka = k0 + 8 * hl + e;
      int kb = k0 + 16 + 8 * hl + e;
      float xa = (ka < H) ? row[ka] * wp[ka] : 0.f;
      float xb = (kb < H) ? row[kb] * wp[kb] : 0.f;
      u.v[e]     = (bf16_t)xa;
      u.v[8 + e] = (bf16_t)xb;
    }
    a4[kk] = u.v;
  }
  float n1v[8], mx[8], sm[8];
#pragma unroll
  for (int r = 0; r < 8; ++r) {
    n1v[r] = n1[(b * P + p) * S + i0 + r + 8 * hl];
    mx[r] = -3.0e38f;
    sm[r] = 0.f;
  }
  const bf16_t* Bb = v2_16 + (size_t)b * S * HPAD;
  for (int jt = 0; jt < 16; ++jt) {
    int j0 = jt * 16;
    v8f c = {0.f, 0.f, 0.f, 0.f, 0.f, 0.f, 0.f, 0.f};
#pragma unroll
    for (int kk = 0; kk < 4; ++kk) {
      v16bf bb = load_b(Bb, HPAD, j0 + m, kk * 32, hl);
      c = wmma_bf16(a4[kk], bb, c);
    }
    int j = j0 + m;
    float n2v = n2[(b * P + p) * S + j];
    float pad = (1.f - (float)mask2[b * S + j]) * MINV;
#pragma unroll
    for (int r = 0; r < 8; ++r) {
      float pw = c[r] * frcp(fmaxf(n1v[r] * n2v, EPSV));
      mx[r] = fmaxf(mx[r], pw + pad);
      sm[r] += pw;
    }
  }
#pragma unroll
  for (int d = 1; d < 16; d <<= 1) {
#pragma unroll
    for (int r = 0; r < 8; ++r) {
      mx[r] = fmaxf(mx[r], __shfl_xor(mx[r], d, 32));
      sm[r] += __shfl_xor(sm[r], d, 32);
    }
  }
  if (m == 0) {
    float rL = frcp(fmaxf(len2[b], EPSV));
#pragma unroll
    for (int r = 0; r < 8; ++r) {
      int i = i0 + r + 8 * hl;
      float* o = out + ((size_t)(outSide * BB + b) * S + i) * OD;
      o[23 + p] = mx[r];
      o[43 + p] = sm[r] * rL;
    }
  }
}

// ---------------------------------------------------------------- att means
// grid(ceil(H/16), S/16, B), block 32. D(i,h) = att(i,:) @ ctxT(h,:) * rsum(i)
__global__ void k_attmean(const bf16_t* A16, const bf16_t* Bt16,
                          const float* rsums, float* outmean) {
  int ht = blockIdx.x, it = blockIdx.y, b = blockIdx.z;
  int lane = threadIdx.x, m = lane & 15, hl = lane >> 4;
  int i0 = it * 16, h0 = ht * 16;
  const bf16_t* A  = A16 + (size_t)b * S * S;
  const bf16_t* Bm = Bt16 + (size_t)b * HPAD * S;
  v8f c = {0.f, 0.f, 0.f, 0.f, 0.f, 0.f, 0.f, 0.f};
  for (int k0 = 0; k0 < S; k0 += 32) {
    v16bf a  = load_a(A, S, i0 + m, k0, hl);
    v16bf bb = load_b(Bm, S, h0 + m, k0, hl);
    c = wmma_bf16(a, bb, c);
  }
  int h = h0 + m;
  if (h < H) {
#pragma unroll
    for (int r = 0; r < 8; ++r) {
      int i = i0 + r + 8 * hl;
      outmean[((size_t)b * S + i) * H + h] = c[r] * rsums[b * S + i];
    }
  }
}

// grid(S, B), block 128. Masked element-wise max of ctx[u,h]*att(s,u).
__global__ void k_attmax(const float* ctx, const float* cosm, const int* msk,
                         float* outmax, int side) {
  int s = blockIdx.x, b = blockIdx.y, t = threadIdx.x;
  __shared__ float satt[S], spad[S];
  for (int u = t; u < S; u += 128) {
    satt[u] = (side == 0) ? cosm[((size_t)b * S + s) * S + u]
                          : cosm[((size_t)b * S + u) * S + s];
    spad[u] = (1.f - (float)msk[b * S + u]) * MINV;
  }
  __syncthreads();
  if (t < H) {
    float mxv = -3.0e38f;
    for (int u = 0; u < S; ++u)
      mxv = fmaxf(mxv, ctx[((size_t)b * S + u) * H + t] * satt[u] + spad[u]);
    outmax[((size_t)b * S + s) * H + t] = mxv;
  }
}

// ---------------------------------------------------------------- final mpm
// grid(S, B, 2), block 128. full / attentive / max-attentive matches:
// unweighted cosine + 20 perspective cosines each.
__global__ void k_mpm_final(const float* cp, const float* ch,
                            const float* last_p, const float* last_h,
                            const float* amean_h, const float* amean_p,
                            const float* amax_h, const float* amax_p,
                            const float* w_full, const float* w_att,
                            const float* w_matt, float* out) {
  int s = blockIdx.x, b = blockIdx.y, side = blockIdx.z, t = threadIdx.x;
  const float* v1  = (side ? ch : cp) + ((size_t)b * S + s) * H;
  const float* v2a = (side ? last_p : last_h) + b * H;
  const float* v2b = (side ? amean_p : amean_h) + ((size_t)b * S + s) * H;
  const float* v2c = (side ? amax_p : amax_h) + ((size_t)b * S + s) * H;
  __shared__ float s1[H], s2[3][H], red[128];
  if (t < H) { s1[t] = v1[t]; s2[0][t] = v2a[t]; s2[1][t] = v2b[t]; s2[2][t] = v2c[t]; }
  __syncthreads();
  red[t] = (t < H) ? s1[t] * s1[t] : 0.f;
  __syncthreads();
  for (int w = 64; w > 0; w >>= 1) { if (t < w) red[t] += red[t + w]; __syncthreads(); }
  float nn1 = fmaxf(sqrtf(red[0]), EPSV);
  __syncthreads();
  float* o = out + ((size_t)(side * BB + b) * S + s) * OD;
  const float* Ws[3] = {w_full, w_att, w_matt};
  const int baseOne[3] = {2, 63, 84};
  for (int mm = 0; mm < 3; ++mm) {
    red[t] = (t < H) ? s1[t] * s2[mm][t] : 0.f;
    __syncthreads();
    for (int w = 64; w > 0; w >>= 1) { if (t < w) red[t] += red[t + w]; __syncthreads(); }
    float dot = red[0];
    __syncthreads();
    red[t] = (t < H) ? s2[mm][t] * s2[mm][t] : 0.f;
    __syncthreads();
    for (int w = 64; w > 0; w >>= 1) { if (t < w) red[t] += red[t + w]; __syncthreads(); }
    float nn2 = fmaxf(sqrtf(red[0]), EPSV);
    __syncthreads();
    if (t == 0) o[baseOne[mm]] = dot * frcp(nn1 * nn2);
    if (t < P) {
      const float* wr = Ws[mm] + t * H;
      float dw = 0.f, a1 = 0.f, a2 = 0.f;
      for (int h = 0; h < H; ++h) {
        float ww = wr[h] * wr[h];
        dw += ww * s1[h] * s2[mm][h];
        a1 += ww * s1[h] * s1[h];
        a2 += ww * s2[mm][h] * s2[mm][h];
      }
      o[baseOne[mm] + 1 + t] =
          dw * frcp(fmaxf(sqrtf(a1), EPSV) * fmaxf(sqrtf(a2), EPSV));
    }
    __syncthreads();
  }
}

// ---------------------------------------------------------------- launch
extern "C" void kernel_launch(void* const* d_in, const int* in_sizes, int n_in,
                              void* d_out, int out_size, void* d_ws,
                              size_t ws_size, hipStream_t stream) {
  (void)in_sizes; (void)n_in; (void)out_size;
  const float* ctx_p  = (const float*)d_in[0];
  const int*   mask_p = (const int*)d_in[1];
  const float* ctx_h  = (const float*)d_in[2];
  const int*   mask_h = (const int*)d_in[3];
  const float* w_full = (const float*)d_in[4];
  const float* w_mp   = (const float*)d_in[5];
  const float* w_att  = (const float*)d_in[6];
  const float* w_matt = (const float*)d_in[7];
  float* out = (float*)d_out;

  float* F = (float*)d_ws;
  size_t off = 0;
  auto alloc = [&](size_t n) { float* p = F + off; off += (n + 63) & ~(size_t)63; return p; };
  float* cp      = alloc((size_t)BB * S * H);
  float* ch      = alloc((size_t)BB * S * H);
  float* rnorm_p = alloc(BB * S);
  float* rnorm_h = alloc(BB * S);
  float* len_p   = alloc(BB);
  float* len_h   = alloc(BB);
  float* last_p  = alloc(BB * H);
  float* last_h  = alloc(BB * H);
  float* w2mp    = alloc(P * HPAD);
  float* nwp     = alloc((size_t)BB * P * S);
  float* nwh     = alloc((size_t)BB * P * S);
  float* cosm    = alloc((size_t)BB * S * S);
  float* rrowsum = alloc(BB * S);
  float* rcolsum = alloc(BB * S);
  float* amean_h = alloc((size_t)BB * S * H);
  float* amean_p = alloc((size_t)BB * S * H);
  float* amax_h  = alloc((size_t)BB * S * H);
  float* amax_p  = alloc((size_t)BB * S * H);

  bf16_t* Bbase = (bf16_t*)(F + off);
  size_t boff = 0;
  auto balloc = [&](size_t n) { bf16_t* p = Bbase + boff; boff += (n + 127) & ~(size_t)127; return p; };
  bf16_t* cp16   = balloc((size_t)BB * S * HPAD);
  bf16_t* ch16   = balloc((size_t)BB * S * HPAD);
  bf16_t* cpT16  = balloc((size_t)BB * HPAD * S);
  bf16_t* chT16  = balloc((size_t)BB * HPAD * S);
  bf16_t* att16  = balloc((size_t)BB * S * S);
  bf16_t* attT16 = balloc((size_t)BB * S * S);

  size_t needed = off * sizeof(float) + boff * sizeof(bf16_t);
  if (ws_size < needed) return;

  k_prep<<<dim3(S, BB, 2), 128, 0, stream>>>(ctx_p, mask_p, ctx_h, mask_h, cp, ch,
                                             cp16, ch16, cpT16, chT16,
                                             rnorm_p, rnorm_h);
  k_lens_last<<<dim3(BB, 2), 256, 0, stream>>>(mask_p, mask_h, cp, ch,
                                               len_p, len_h, last_p, last_h);
  k_w2<<<dim3(P), 128, 0, stream>>>(w_mp, w2mp);
  k_wnorm<<<dim3(S, BB, 2), 32, 0, stream>>>(cp, ch, w2mp, nwp, nwh);
  k_cos_gemm<<<dim3(S / 16, S / 16, BB), 32, 0, stream>>>(cp16, ch16, rnorm_p,
                                                          rnorm_h, cosm);
  k_reduce_rows<<<dim3(S, BB), 256, 0, stream>>>(cosm, mask_h, len_h, att16,
                                                 attT16, rrowsum, out);
  k_reduce_cols<<<dim3(S, BB), 256, 0, stream>>>(cosm, mask_p, len_p, rcolsum, out);
  k_pw<<<dim3(S / 16, P, BB), 32, 0, stream>>>(cp, ch16, w2mp, nwp, nwh, mask_h,
                                               len_h, out, 0);
  k_pw<<<dim3(S / 16, P, BB), 32, 0, stream>>>(ch, cp16, w2mp, nwh, nwp, mask_p,
                                               len_p, out, 1);
  k_attmean<<<dim3((H + 15) / 16, S / 16, BB), 32, 0, stream>>>(att16, chT16,
                                                                rrowsum, amean_h);
  k_attmean<<<dim3((H + 15) / 16, S / 16, BB), 32, 0, stream>>>(attT16, cpT16,
                                                                rcolsum, amean_p);
  k_attmax<<<dim3(S, BB), 128, 0, stream>>>(ch, cosm, mask_h, amax_h, 0);
  k_attmax<<<dim3(S, BB), 128, 0, stream>>>(cp, cosm, mask_p, amax_p, 1);
  k_mpm_final<<<dim3(S, BB, 2), 128, 0, stream>>>(cp, ch, last_p, last_h, amean_h,
                                                  amean_p, amax_h, amax_p, w_full,
                                                  w_att, w_matt, out);
}
